// PathSwappedEncoder_16234976378939
// MI455X (gfx1250) — matrix operand order repaired and checked by reading, not compiled, not run on hardware
//
#include <hip/hip_runtime.h>
#include <hip/hip_bf16.h>
#include <math.h>

// ---------------------------------------------------------------------------
// Problem constants (from reference): B=256, N=129, C=768
// ---------------------------------------------------------------------------
#define B_DIM   256
#define N_DIM   129
#define C_DIM   768
#define H_DIM   192      // C/4
#define G_DIM   256
#define TOKENS  (B_DIM * N_DIM)        // 33024
#define NSWAPP  38                     // int(129 * 0.3)
#define NUM_SWAP 2

typedef __bf16 bf16;
typedef bf16  v16bf __attribute__((ext_vector_type(16)));
typedef bf16  v8bf  __attribute__((ext_vector_type(8)));
typedef float v8f   __attribute__((ext_vector_type(8)));

// f32 -> bf16 with round-to-nearest-even
__device__ __forceinline__ bf16 f2bf(float f) {
  unsigned u = __builtin_bit_cast(unsigned, f);
  unsigned r = u + 0x7FFFu + ((u >> 16) & 1u);
  unsigned short h = (unsigned short)(r >> 16);
  return __builtin_bit_cast(bf16, h);
}
__device__ __forceinline__ float bf2f(bf16 b) {
  unsigned short h = __builtin_bit_cast(unsigned short, b);
  unsigned u = ((unsigned)h) << 16;
  return __builtin_bit_cast(float, u);
}

// A-fragment loader (16x32 bf16, contiguous k order, matches B):
// lane<16 : row M=lane,   elems 0..15 -> K=k0+0..15
// lane>=16: row M=lane-16,elems 0..15 -> K=k0+16..31
__device__ __forceinline__ v16bf load_a(const bf16* rowbase, int k0, int abase) {
  return *(const v16bf*)(rowbase + k0 + 2 * abase);
}

// ---------------------------------------------------------------------------
// K_pre: convert + transpose weights to bf16 [n][k] so B-fragments are
// contiguous 32B per-lane loads.
// ---------------------------------------------------------------------------
__global__ void convert_weights(const float* __restrict__ W1,
                                const float* __restrict__ W2,
                                const float* __restrict__ V1,
                                bf16* __restrict__ W1t,   // [192][768]
                                bf16* __restrict__ W2t,   // [768][192]
                                bf16* __restrict__ V1t)   // [256][768]
{
  int i = blockIdx.x * blockDim.x + threadIdx.x;
  if (i < H_DIM * C_DIM) {                 // W1 (768x192) -> W1t
    int n = i / C_DIM, k = i % C_DIM;
    W1t[i] = f2bf(W1[k * H_DIM + n]);
  }
  int j = i - H_DIM * C_DIM;
  if (j >= 0 && j < C_DIM * H_DIM) {       // W2 (192x768) -> W2t
    int n = j / H_DIM, k = j % H_DIM;
    W2t[j] = f2bf(W2[k * C_DIM + n]);
  }
  int l = i - 2 * H_DIM * C_DIM;
  if (l >= 0 && l < G_DIM * C_DIM) {       // V1 (768x256) -> V1t
    int n = l / C_DIM, k = l % C_DIM;
    V1t[l] = f2bf(V1[k * G_DIM + n]);
  }
}

// ---------------------------------------------------------------------------
// K0: cls_n[b] = features[b,0,:] / max(||.||, 1e-12)
// ---------------------------------------------------------------------------
__global__ void cls_normalize(const float* __restrict__ features,
                              float* __restrict__ cls_n)
{
  __shared__ float red[256];
  int b = blockIdx.x, t = threadIdx.x;
  const float* row = features + (size_t)b * N_DIM * C_DIM;
  float s = 0.f;
  for (int c = t; c < C_DIM; c += 256) { float v = row[c]; s += v * v; }
  red[t] = s; __syncthreads();
  for (int off = 128; off > 0; off >>= 1) {
    if (t < off) red[t] += red[t + off];
    __syncthreads();
  }
  float denom = fmaxf(sqrtf(red[0]), 1e-12f);
  for (int c = t; c < C_DIM; c += 256) cls_n[b * C_DIM + c] = row[c] / denom;
}

// ---------------------------------------------------------------------------
// K1: sim row + same-label mask + top-4; keep ranks 2,3 as swap_indices.
// ---------------------------------------------------------------------------
__global__ void sim_topk(const float* __restrict__ cls_n,
                         const int* __restrict__ labels,
                         int* __restrict__ swap_indices)
{
  __shared__ float myrow[C_DIM];
  __shared__ float simrow[B_DIM];
  int b = blockIdx.x, t = threadIdx.x;   // blockDim = 256
  for (int c = t; c < C_DIM; c += 256) myrow[c] = cls_n[b * C_DIM + c];
  __syncthreads();
  const float* other = cls_n + (size_t)t * C_DIM;
  float dot = 0.f;
  for (int c = 0; c < C_DIM; ++c) dot += myrow[c] * other[c];
  if (labels[t] == labels[b]) dot = -__builtin_inff();
  simrow[t] = dot;
  __syncthreads();
  if (t == 0) {
    for (int pass = 0; pass < 4; ++pass) {   // k = NUM_SWAP + 2 = 4
      int best = 0; float bv = simrow[0];
      for (int j = 1; j < B_DIM; ++j)
        if (simrow[j] > bv) { bv = simrow[j]; best = j; }
      simrow[best] = -__builtin_inff();
      if (pass >= 2) swap_indices[b * NUM_SWAP + (pass - 2)] = best;
    }
  }
}

// ---------------------------------------------------------------------------
// K2: fused importance MLP per 16-token tile, one wave32 per block.
// imp = relu(relu(X@W1+b1)@W2+b2 @ V1 + c1) @ V2 + c2  (softmax skipped:
// monotonic, only used for top-k).  All GEMMs via v_wmma_f32_16x16x32_bf16.
// ---------------------------------------------------------------------------
__global__ void __launch_bounds__(32)
importance_mlp(const float* __restrict__ features,
               const bf16* __restrict__ W1t, const float* __restrict__ b1,
               const bf16* __restrict__ W2t, const float* __restrict__ b2,
               const bf16* __restrict__ V1t, const float* __restrict__ c1,
               const float* __restrict__ V2, const float* __restrict__ c2,
               float* __restrict__ imp)
{
  __shared__ bf16 bufX[16 * C_DIM];    // X tile, later reused for H2
  __shared__ bf16 bufH1[16 * H_DIM];
  __shared__ bf16 bufG[16 * G_DIM];

  const int lane = threadIdx.x;
  const int half = lane >> 4;          // 0 or 1
  const int lm   = lane & 15;
  const int abase = half ? 8 : 0;
  const size_t tok0 = (size_t)blockIdx.x * 16;

  // stage X (f32 -> bf16) into LDS
  const float* X = features + tok0 * C_DIM;
  for (int i = lane; i < 16 * C_DIM; i += 32) bufX[i] = f2bf(X[i]);
  __syncthreads();

  // ---- GEMM1: H1 = relu(X @ W1 + b1)   M=16 K=768 N=192 ----
  for (int n0 = 0; n0 < H_DIM; n0 += 16) {
    v8f acc = {};
    const bf16* bptr = W1t + (size_t)(n0 + lm) * C_DIM + half * 16;
    for (int k0 = 0; k0 < C_DIM; k0 += 32) {
      __builtin_prefetch(bptr + k0 + 128, 0, 0);
      v16bf a  = load_a(&bufX[lm * C_DIM], k0, abase);
      v16bf bb = *(const v16bf*)(bptr + k0);
      acc = __builtin_amdgcn_wmma_f32_16x16x32_bf16(false, a, false, bb,
                                                    (short)0, acc, false, false);
    }
    float bias = b1[n0 + lm];
    for (int r = 0; r < 8; ++r) {
      int m = r + 8 * half;
      bufH1[m * H_DIM + n0 + lm] = f2bf(fmaxf(acc[r] + bias, 0.f));
    }
  }
  __syncthreads();

  // ---- GEMM2: H2 = H1 @ W2 + b2   M=16 K=192 N=768 (into bufX) ----
  for (int n0 = 0; n0 < C_DIM; n0 += 16) {
    v8f acc = {};
    const bf16* bptr = W2t + (size_t)(n0 + lm) * H_DIM + half * 16;
    for (int k0 = 0; k0 < H_DIM; k0 += 32) {
      v16bf a  = load_a(&bufH1[lm * H_DIM], k0, abase);
      v16bf bb = *(const v16bf*)(bptr + k0);
      acc = __builtin_amdgcn_wmma_f32_16x16x32_bf16(false, a, false, bb,
                                                    (short)0, acc, false, false);
    }
    float bias = b2[n0 + lm];
    for (int r = 0; r < 8; ++r) {
      int m = r + 8 * half;
      bufX[m * C_DIM + n0 + lm] = f2bf(acc[r] + bias);
    }
  }
  __syncthreads();

  // ---- GEMM3: G = relu(H2 @ V1 + c1)   M=16 K=768 N=256 ----
  for (int n0 = 0; n0 < G_DIM; n0 += 16) {
    v8f acc = {};
    const bf16* bptr = V1t + (size_t)(n0 + lm) * C_DIM + half * 16;
    for (int k0 = 0; k0 < C_DIM; k0 += 32) {
      __builtin_prefetch(bptr + k0 + 128, 0, 0);
      v16bf a  = load_a(&bufX[lm * C_DIM], k0, abase);
      v16bf bb = *(const v16bf*)(bptr + k0);
      acc = __builtin_amdgcn_wmma_f32_16x16x32_bf16(false, a, false, bb,
                                                    (short)0, acc, false, false);
    }
    float bias = c1[n0 + lm];
    for (int r = 0; r < 8; ++r) {
      int m = r + 8 * half;
      bufG[m * G_DIM + n0 + lm] = f2bf(fmaxf(acc[r] + bias, 0.f));
    }
  }
  __syncthreads();

  // ---- imp = G @ V2 + c2 (N=1): VALU dot per token ----
  if (lane < 16) {
    float s = c2[0];
    for (int k = 0; k < G_DIM; ++k)
      s += bf2f(bufG[lane * G_DIM + k]) * V2[k];
    imp[tok0 + lane] = s;
  }
}

// ---------------------------------------------------------------------------
// K3: per batch: top-38 of imp row (ties -> lowest index, matches lax.top_k);
// build src_row[b][patch] = cls row replacing that patch (-1 = keep).
// Sequential j order so duplicate adj indices resolve "last wins".
// ---------------------------------------------------------------------------
__global__ void select_patches(const float* __restrict__ imp,
                               const int* __restrict__ swap_indices,
                               const int* __restrict__ swap_choice,
                               int* __restrict__ src_row)
{
  __shared__ float vals[N_DIM];
  int b = blockIdx.x, t = threadIdx.x;   // blockDim = 128
  if (t < N_DIM - 1) src_row[b * (N_DIM - 1) + t] = -1;
  for (int i = t; i < N_DIM; i += 128) vals[i] = imp[b * N_DIM + i];
  __syncthreads();
  if (t == 0) {
    for (int j = 0; j < NSWAPP; ++j) {
      int best = 0; float bv = vals[0];
      for (int p = 1; p < N_DIM; ++p)
        if (vals[p] > bv) { bv = vals[p]; best = p; }
      vals[best] = -__builtin_inff();
      int adj = best - 1; if (adj < 0) adj = 0;
      int choice = swap_choice[b * NSWAPP + j];
      src_row[b * (N_DIM - 1) + adj] = swap_indices[b * NUM_SWAP + choice];
    }
  }
}

// ---------------------------------------------------------------------------
// K4: out = 0.7*features + 0.3*swapped  (bitwise same formula as reference)
// ---------------------------------------------------------------------------
__global__ void blend_out(const float* __restrict__ features,
                          const float* __restrict__ cls_n,
                          const int* __restrict__ src_row,
                          float* __restrict__ out)
{
  int row = blockIdx.x;                  // b*129 + n
  int b = row / N_DIM, n = row % N_DIM;
  int t = threadIdx.x;                   // 256
  const float* f = features + (size_t)row * C_DIM;
  float* o = out + (size_t)row * C_DIM;
  int r = -1;
  if (n > 0) r = src_row[b * (N_DIM - 1) + (n - 1)];
  const float* s = (r >= 0) ? (cls_n + (size_t)r * C_DIM) : f;
  for (int c = t; c < C_DIM; c += 256)
    o[c] = f[c] * 0.7f + s[c] * 0.3f;
}

// ---------------------------------------------------------------------------
// launch
// ---------------------------------------------------------------------------
extern "C" void kernel_launch(void* const* d_in, const int* in_sizes, int n_in,
                              void* d_out, int out_size, void* d_ws, size_t ws_size,
                              hipStream_t stream) {
  const float* features    = (const float*)d_in[0];
  const float* W1          = (const float*)d_in[1];
  const float* b1          = (const float*)d_in[2];
  const float* W2          = (const float*)d_in[3];
  const float* b2          = (const float*)d_in[4];
  const float* V1          = (const float*)d_in[5];
  const float* c1          = (const float*)d_in[6];
  const float* V2          = (const float*)d_in[7];
  const float* c2          = (const float*)d_in[8];
  const int*   labels      = (const int*)d_in[9];
  const int*   swap_choice = (const int*)d_in[10];
  float* out = (float*)d_out;

  char* ws = (char*)d_ws;
  float* cls_n     = (float*)(ws + 0);            // 256*768*4      = 786432
  bf16*  W1t       = (bf16*)(ws + 786432);        // 192*768*2      = 294912
  bf16*  W2t       = (bf16*)(ws + 1081344);       // 768*192*2      = 294912
  bf16*  V1t       = (bf16*)(ws + 1376256);       // 256*768*2      = 393216
  float* imp       = (float*)(ws + 1769472);      // 33024*4        = 132096
  int*   swap_idx  = (int*)(ws + 1901568);        // 256*2*4        = 2048
  int*   src_row   = (int*)(ws + 1903616);        // 256*128*4      = 131072

  convert_weights<<<(2 * H_DIM * C_DIM + G_DIM * C_DIM + 255) / 256, 256, 0, stream>>>(
      W1, W2, V1, W1t, W2t, V1t);
  cls_normalize<<<B_DIM, 256, 0, stream>>>(features, cls_n);
  importance_mlp<<<TOKENS / 16, 32, 0, stream>>>(features, W1t, b1, W2t, b2,
                                                 V1t, c1, V2, c2, imp);
  sim_topk<<<B_DIM, 256, 0, stream>>>(cls_n, labels, swap_idx);
  select_patches<<<B_DIM, 128, 0, stream>>>(imp, swap_idx, swap_choice, src_row);
  blend_out<<<TOKENS, 256, 0, stream>>>(features, cls_n, src_row, out);
}
